// DGN_48387101557080
// MI455X (gfx1250) — matrix operand order repaired and checked by reading, not compiled, not run on hardware
//
#include <hip/hip_runtime.h>
#include <hip/hip_bf16.h>

typedef __attribute__((ext_vector_type(16))) _Float16 v16h;
typedef __attribute__((ext_vector_type(8)))  float    v8f;

#define NN 2048
#define EE 65536
#define NVIEW 6

// --- degree counts (shared by all 3 layers) ------------------------------
__global__ void count_kernel(const int* __restrict__ ei, float* __restrict__ cnt) {
    int t = blockIdx.x * blockDim.x + threadIdx.x;
    if (t < EE) atomicAdd(&cnt[ei[EE + t]], 1.0f);
}

// --- fused edge kernel: edge-MLP (WMMA) -> per-edge matvec -> scatter-add --
// 256 threads (8 waves) handle 16 edges.
// Edge MLP GEMM: A = edge_attr tile [16 x 6] zero-padded to K=32 (f16),
//                B = W [6 x cols] zero-padded (f16), cols = in_c*32.
// One v_wmma_f32_16x16x32_f16 per 16-column N-tile; bias+relu into LDS.
__global__ __launch_bounds__(256)
void edge_kernel(const float* __restrict__ edge_attr,  // [E, 6]
                 const int*   __restrict__ ei,         // [2, E]
                 const float* __restrict__ W,          // [6, cols]
                 const float* __restrict__ bvec,       // [cols]
                 const float* __restrict__ h_in,       // [N, in_c]
                 float*       __restrict__ accum,      // [N, 32]
                 int in_c)
{
    const int cols = in_c * 32;
    __shared__ _Float16 wtile[16 * 1024];   // 32 KB max (in_c=32)
    __shared__ float    hs[16 * 32];
    __shared__ int      lsrc[16], ldst[16];

    const int t     = threadIdx.x;
    const int lane  = t & 31;
    const int wave  = t >> 5;
    const int ebase = blockIdx.x * 16;

    // gfx1250 prefetch of this block's edge_attr tile (global_prefetch_b8)
    __builtin_prefetch(edge_attr + (size_t)(ebase + (lane & 15)) * NVIEW, 0, 1);

    if (t < 16) {
        lsrc[t] = ei[ebase + t];
        ldst[t] = ei[EE + ebase + t];
    }

    // A fragment: 16-bit A 16x32 layout; lanes 0-15 = rows (edges),
    // v16h elements 0..7 = K0..K7 (K>=6 stays zero), lanes 16-31 = K8.. (zero).
    v16h afrag = {};
    if (lane < 16) {
        const float* ap = edge_attr + (size_t)(ebase + lane) * NVIEW;
#pragma unroll
        for (int k = 0; k < NVIEW; ++k) afrag[k] = (_Float16)ap[k];
    }

    const int ntiles = cols >> 4;
    for (int tile = wave; tile < ntiles; tile += 8) {   // wave-uniform branch
        const int nbase = tile << 4;
        // B fragment: dense B 32x16; lanes 0-15 hold K=0..7,16..23 for col = lane.
        // Only K<6 is non-zero (A is zero-padded beyond).
        v16h bfrag = {};
        if (lane < 16) {
            const int col = nbase + lane;
#pragma unroll
            for (int k = 0; k < NVIEW; ++k) bfrag[k] = (_Float16)W[k * cols + col];
        }
        v8f c = {};
        c = __builtin_amdgcn_wmma_f32_16x16x32_f16(false, afrag, false, bfrag,
                                                   (short)0, c, false, false);
        // C/D layout: vgpr r, lanes 0-15 -> M=r, lanes 16-31 -> M=r+8; N = lane&15
        const int  ncol = nbase + (lane & 15);
        const int  mofs = (lane >= 16) ? 8 : 0;
        const float bb  = bvec[ncol];
#pragma unroll
        for (int r = 0; r < 8; ++r) {
            float v = c[r] + bb;
            wtile[(r + mofs) * cols + ncol] = (_Float16)(v > 0.0f ? v : 0.0f);
        }
    }
    __syncthreads();

    // stage h[src] rows for the 16 edges
    for (int idx = t; idx < 16 * in_c; idx += 256) {
        int e = idx / in_c, i = idx - e * in_c;
        hs[e * in_c + i] = h_in[(size_t)lsrc[e] * in_c + i];
    }
    __syncthreads();

    // per-edge matvec: msg[e,o] = sum_i h[src_e,i] * w_e[i,o]; scatter to dst
    for (int task = t; task < 16 * 32; task += 256) {
        const int e = task >> 5, o = task & 31;
        float acc = 0.0f;
        for (int i = 0; i < in_c; ++i)
            acc += hs[e * in_c + i] * (float)wtile[e * cols + i * 32 + o];
        atomicAdd(&accum[(size_t)ldst[e] * 32 + o], acc);
    }
}

// --- node update, layer 1 (in_c = 1): VALU path ---------------------------
__global__ void node_kernel1(const float* __restrict__ accum,
                             const float* __restrict__ cnt,
                             const float* __restrict__ h_in,   // [N, 1]
                             const float* __restrict__ root,   // [1, 32]
                             const float* __restrict__ bias,   // [32]
                             float*       __restrict__ h_out)  // [N, 32]
{
    int t = blockIdx.x * blockDim.x + threadIdx.x;
    if (t >= NN * 32) return;
    int n = t >> 5, o = t & 31;
    float cc = cnt[n]; cc = cc > 1.0f ? cc : 1.0f;
    float v = accum[t] / cc + h_in[n] * root[o] + bias[o];
    h_out[t] = v > 0.0f ? v : 0.0f;
}

// --- node update, layers 2/3 (in_c = 32): root GEMM on WMMA ---------------
// One wave per 16-node tile; K = 32 -> exactly one v_wmma_f32_16x16x32_f16
// per 16-column N-tile (2 tiles). Epilogue fuses mean-agg + bias + relu.
__global__ __launch_bounds__(32)
void node_wmma_kernel(const float* __restrict__ accum,  // [N, 32]
                      const float* __restrict__ cnt,    // [N]
                      const float* __restrict__ h_in,   // [N, 32]
                      const float* __restrict__ root,   // [32, 32]
                      const float* __restrict__ bias,   // [32]
                      float*       __restrict__ h_out)  // [N, 32]
{
    const int lane  = threadIdx.x & 31;
    const int nbase = blockIdx.x * 16;
    const int row   = nbase + (lane & 15);

    // A fragment (16-bit A 16x32): lane l holds row (l&15);
    // elements 0..7 -> K = kb..kb+7, elements 8..15 -> K = kb+16..kb+23,
    // kb = (l>=16) ? 8 : 0.
    const int kb = (lane >= 16) ? 8 : 0;
    v16h afrag;
#pragma unroll
    for (int j = 0; j < 8; ++j) {
        afrag[j]     = (_Float16)h_in[(size_t)row * 32 + kb + j];
        afrag[j + 8] = (_Float16)h_in[(size_t)row * 32 + kb + 16 + j];
    }

    // B fragment (dense B 32x16): lane l holds col (l&15);
    // element j -> K = kbb + j + (j>=8 ? 8 : 0), kbb = (l>=16) ? 8 : 0.
    const int kbb  = (lane >= 16) ? 8 : 0;
    const int colb = lane & 15;
    const int mofs = (lane >= 16) ? 8 : 0;

#pragma unroll
    for (int nt = 0; nt < 2; ++nt) {
        const int col = nt * 16 + colb;
        v16h bfrag;
#pragma unroll
        for (int j = 0; j < 16; ++j)
            bfrag[j] = (_Float16)root[(size_t)(kbb + j + (j >= 8 ? 8 : 0)) * 32 + col];
        v8f c = {};
        c = __builtin_amdgcn_wmma_f32_16x16x32_f16(false, afrag, false, bfrag,
                                                   (short)0, c, false, false);
        const float bb = bias[col];
#pragma unroll
        for (int r = 0; r < 8; ++r) {
            const int n = nbase + r + mofs;
            float cc = cnt[n]; cc = cc > 1.0f ? cc : 1.0f;
            float v = c[r] + accum[(size_t)n * 32 + col] / cc + bb;
            h_out[(size_t)n * 32 + col] = v > 0.0f ? v : 0.0f;
        }
    }
}

// --- all-pairs L1 distance (CBT), 32x32 output tile per block -------------
__global__ __launch_bounds__(256)
void cbt_kernel(const float* __restrict__ h, float* __restrict__ out)
{
    __shared__ float hi[32][33];   // pad: stride 33 avoids 32-bank conflicts
    __shared__ float hj[32][33];
    const int t = threadIdx.x;
    const int ibase = blockIdx.y * 32, jbase = blockIdx.x * 32;
    for (int idx = t; idx < 1024; idx += 256) {
        int r = idx >> 5, k = idx & 31;
        hi[r][k] = h[(size_t)(ibase + r) * 32 + k];
        hj[r][k] = h[(size_t)(jbase + r) * 32 + k];
    }
    __syncthreads();
    const int jl = t & 31;
    const int r0 = t >> 5;
#pragma unroll
    for (int rr = 0; rr < 4; ++rr) {
        const int il = r0 + rr * 8;
        float s = 0.0f;
#pragma unroll
        for (int k = 0; k < 32; ++k) s += fabsf(hj[jl][k] - hi[il][k]);
        out[(size_t)(ibase + il) * NN + (jbase + jl)] = s;
    }
}

extern "C" void kernel_launch(void* const* d_in, const int* in_sizes, int n_in,
                              void* d_out, int out_size, void* d_ws, size_t ws_size,
                              hipStream_t stream)
{
    const float* x         = (const float*)d_in[0];
    const float* edge_attr = (const float*)d_in[1];
    const int*   ei        = (const int*)d_in[2];
    const float* W1 = (const float*)d_in[3];
    const float* b1 = (const float*)d_in[4];
    const float* root1 = (const float*)d_in[5];
    const float* bias1 = (const float*)d_in[6];
    const float* W2 = (const float*)d_in[7];
    const float* b2 = (const float*)d_in[8];
    const float* root2 = (const float*)d_in[9];
    const float* bias2 = (const float*)d_in[10];
    const float* W3 = (const float*)d_in[11];
    const float* b3 = (const float*)d_in[12];
    const float* root3 = (const float*)d_in[13];
    const float* bias3 = (const float*)d_in[14];

    float* ws    = (float*)d_ws;
    float* cnt   = ws;                       // 2048
    float* accum = ws + 2048;                // 65536
    float* h1    = accum + 65536;            // 65536
    float* h2    = h1 + 65536;               // 65536
    float* h3    = h2 + 65536;               // 65536  (total ~1.06 MB)

    hipMemsetAsync(cnt, 0, NN * sizeof(float), stream);
    count_kernel<<<EE / 256, 256, 0, stream>>>(ei, cnt);

    // layer 1 (in_c = 1)
    hipMemsetAsync(accum, 0, NN * 32 * sizeof(float), stream);
    edge_kernel<<<EE / 16, 256, 0, stream>>>(edge_attr, ei, W1, b1, x, accum, 1);
    node_kernel1<<<NN * 32 / 256, 256, 0, stream>>>(accum, cnt, x, root1, bias1, h1);

    // layer 2 (in_c = 32)
    hipMemsetAsync(accum, 0, NN * 32 * sizeof(float), stream);
    edge_kernel<<<EE / 16, 256, 0, stream>>>(edge_attr, ei, W2, b2, h1, accum, 32);
    node_wmma_kernel<<<NN / 16, 32, 0, stream>>>(accum, cnt, h1, root2, bias2, h2);

    // layer 3 (in_c = 32)
    hipMemsetAsync(accum, 0, NN * 32 * sizeof(float), stream);
    edge_kernel<<<EE / 16, 256, 0, stream>>>(edge_attr, ei, W3, b3, h2, accum, 32);
    node_wmma_kernel<<<NN / 16, 32, 0, stream>>>(accum, cnt, h2, root3, bias3, h3);

    dim3 g(NN / 32, NN / 32);
    cbt_kernel<<<g, 256, 0, stream>>>(h3, (float*)d_out);
}
